// SnippetTopicGCN_31430570672689
// MI455X (gfx1250) — matrix operand order repaired and checked by reading, not compiled, not run on hardware
//
#include <hip/hip_runtime.h>
#include <stdint.h>

typedef __attribute__((ext_vector_type(16))) __bf16 v16bf;
typedef __attribute__((ext_vector_type(8)))  float  v8f;

#define B_N 8
#define T_N 2048

// ---------- helpers ----------
__device__ __forceinline__ unsigned short f2bf(float f) {
  union { float f; unsigned u; } v; v.f = f;
  unsigned r = v.u + 0x7FFFu + ((v.u >> 16) & 1u);
  return (unsigned short)(r >> 16);
}

// ---------- front-end: grouped conv1d k=3 groups=4 (256->256) + relu ----------
// snip is [B][256][T] (input layout); x/xb are written t-major [B][T][256]
__global__ void bb1_kernel(const float* __restrict__ snip, const float* __restrict__ w,
                           const float* __restrict__ bias, float* __restrict__ x,
                           unsigned short* __restrict__ xb) {
  int idx = blockIdx.x * blockDim.x + threadIdx.x;      // B*256*T, t fastest
  int t = idx & (T_N - 1);
  int o = (idx >> 11) & 255;
  int b = idx >> 19;
  const float* in = snip + ((size_t)b * 256 + (size_t)(o >> 6) * 64) * T_N;
  const float* wr = w + (size_t)o * 192;
  float sum = bias[o];
  for (int ic = 0; ic < 64; ++ic) {
    const float* row = in + (size_t)ic * T_N;
    float w0 = wr[ic * 3 + 0], w1 = wr[ic * 3 + 1], w2 = wr[ic * 3 + 2];
    if (t > 0)       sum += w0 * row[t - 1];
    sum += w1 * row[t];
    if (t < T_N - 1) sum += w2 * row[t + 1];
  }
  sum = fmaxf(sum, 0.f);
  size_t ot = ((size_t)b * T_N + t) * 256 + o;          // t-major
  x[ot] = sum;
  xb[ot] = f2bf(sum);
}

// ---------- topic embedding conv (per-batch 256 outputs) + relu ----------
__global__ void topic_kernel(const float* __restrict__ te, const float* __restrict__ w,
                             const float* __restrict__ bias, float* __restrict__ out) {
  int idx = blockIdx.x * blockDim.x + threadIdx.x;
  if (idx >= B_N * 256) return;
  int o = idx & 255, b = idx >> 8;
  int g = o >> 6;
  float s = bias[o];
  for (int ic = 0; ic < 4; ++ic) s += w[o * 4 + ic] * te[b * 16 + g * 4 + ic];
  out[idx] = fmaxf(s, 0.f);
}

// ---------- xx[b][t] = sum_c x^2   (x is t-major: contiguous row) ----------
__global__ void xx_kernel(const float* __restrict__ x, float* __restrict__ xx) {
  int idx = blockIdx.x * blockDim.x + threadIdx.x;      // B*T
  const float4* p = (const float4*)(x + (size_t)idx * 256);
  float s = 0.f;
  for (int c = 0; c < 64; ++c) {
    float4 q = p[c];
    s += q.x * q.x + q.y * q.y + q.z * q.z + q.w * q.w;
  }
  xx[idx] = s;
}

// ---------- fp32 -> bf16 cast (weights) ----------
__global__ void cast_kernel(const float* __restrict__ src, unsigned short* __restrict__ dst, int n) {
  int i = blockIdx.x * blockDim.x + threadIdx.x;
  if (i < n) dst[i] = f2bf(src[i]);
}

// ---------- WMMA bf16 GEMM, all operands contiguous b128 loads ----------
// W: [M][K] bf16 row-major.  X: t-major [B][T][K] bf16.  out: t-major [B][T][M] f32.
// One wave per block; wave owns 4 m-tiles (64 rows) x 16 t-columns.
// mode: 0 = store, 1 = store relu, 2 = max-accumulate into out
__global__ __launch_bounds__(32)
void gemm_bf16_kernel(const unsigned short* __restrict__ Wb,
                      const unsigned short* __restrict__ XbT,
                      const float* __restrict__ bias,
                      float* __restrict__ outT,
                      int M, int K, int mode) {
  const int lane = threadIdx.x;
  const int t0 = blockIdx.x * 16;
  const int mbase = blockIdx.y * 64;
  const int b  = blockIdx.z;
  const int hi = lane >> 4;
  const int ml = lane & 15;
  const int kbA = hi * 8, kbB = hi * 8;
  const unsigned short* X = XbT + ((size_t)b * T_N + (t0 + ml)) * K;

  v8f acc[4];
#pragma unroll
  for (int mt = 0; mt < 4; ++mt)
#pragma unroll
    for (int v = 0; v < 8; ++v) acc[mt][v] = 0.f;

  for (int k0 = 0; k0 < K; k0 += 32) {
    // B tile: two contiguous 8-element K runs for this lane's column -> 2 x b128
    union { v16bf v; uint4 q[2]; } bm;
    bm.q[0] = *(const uint4*)(const void*)(X + k0 + kbB);
    bm.q[1] = *(const uint4*)(const void*)(X + k0 + kbB + 16);
#pragma unroll
    for (int mt = 0; mt < 4; ++mt) {
      // A tile: weight row, two contiguous 8-element runs -> 2 x b128
      union { v16bf v; uint4 q[2]; } a;
      const unsigned short* wrow = Wb + (size_t)(mbase + mt * 16 + ml) * K + k0 + kbA;
      a.q[0] = *(const uint4*)(const void*)(wrow);
      a.q[1] = *(const uint4*)(const void*)(wrow + 16);
      acc[mt] = __builtin_amdgcn_wmma_f32_16x16x32_bf16(false, a.v, false, bm.v,
                                                        (short)0, acc[mt], false, false);
    }
  }
  // D store: rows v..v+7 are contiguous in t-major output -> 2 x float4 per tile
  float* orow = outT + ((size_t)b * T_N + (t0 + ml)) * M + mbase + 8 * hi;
#pragma unroll
  for (int mt = 0; mt < 4; ++mt) {
    const float* bptr = bias + mbase + mt * 16 + 8 * hi;
    float* dst = orow + mt * 16;
    float vals[8];
#pragma unroll
    for (int v = 0; v < 8; ++v) {
      float val = acc[mt][v] + bptr[v];
      if (mode == 1) val = fmaxf(val, 0.f);
      vals[v] = val;
    }
    if (mode == 2) {
      float4 o0 = *(float4*)(dst), o1 = *(float4*)(dst + 4);
      vals[0] = fmaxf(vals[0], o0.x); vals[1] = fmaxf(vals[1], o0.y);
      vals[2] = fmaxf(vals[2], o0.z); vals[3] = fmaxf(vals[3], o0.w);
      vals[4] = fmaxf(vals[4], o1.x); vals[5] = fmaxf(vals[5], o1.y);
      vals[6] = fmaxf(vals[6], o1.z); vals[7] = fmaxf(vals[7], o1.w);
    }
    float4 r0, r1;
    r0.x = vals[0]; r0.y = vals[1]; r0.z = vals[2]; r0.w = vals[3];
    r1.x = vals[4]; r1.y = vals[5]; r1.z = vals[6]; r1.w = vals[7];
    *(float4*)(dst) = r0;
    *(float4*)(dst + 4) = r1;
  }
}

// ---------- KNN: WMMA Gram tiles, all-contiguous operands, per-row top-3 ----------
__global__ __launch_bounds__(32)
void knn_kernel(const unsigned short* __restrict__ XbT,  // t-major [B][T][256] bf16
                const float* __restrict__ xx,            // [B][T]
                const int* __restrict__ seg_lens,        // [B]
                int* __restrict__ idx_out) {             // [B][T][3]
  __shared__ float lv[16][16][3];
  __shared__ int   li[16][16][3];
  const int lane = threadIdx.x;
  const int t0 = blockIdx.x * 16;
  const int b  = blockIdx.y;
  const int hi = lane >> 4;
  const int ml = lane & 15;
  const int kbA = hi * 8, kbB = hi * 8;
  const unsigned short* X = XbT + (size_t)b * T_N * 256;

  // Preload 8 A K-tiles: lane's t-row is contiguous -> 2 x b128 each
  v16bf aT[8];
#pragma unroll
  for (int ks = 0; ks < 8; ++ks) {
    union { v16bf v; uint4 q[2]; } a;
    const unsigned short* base = X + (size_t)(t0 + ml) * 256 + ks * 32 + kbA;
    a.q[0] = *(const uint4*)(const void*)(base);
    a.q[1] = *(const uint4*)(const void*)(base + 16);
    aT[ks] = a.v;
  }
  float xxr[8];
#pragma unroll
  for (int v = 0; v < 8; ++v) xxr[v] = xx[b * T_N + t0 + v + 8 * hi];
  const int seg = seg_lens[b];

  float bv0[8], bv1[8], bv2[8]; int bi0[8], bi1[8], bi2[8];
#pragma unroll
  for (int v = 0; v < 8; ++v) { bv0[v] = bv1[v] = bv2[v] = -3.4e38f; bi0[v] = bi1[v] = bi2[v] = 0; }

  for (int s0 = 0; s0 < T_N; s0 += 16) {
    const unsigned short* brow = X + (size_t)(s0 + ml) * 256 + kbB;
    __builtin_prefetch((const void*)(brow + 16 * 256), 0, 1);   // next s-tile
    v8f acc = {};
#pragma unroll
    for (int ks = 0; ks < 8; ++ks) {
      union { v16bf v; uint4 q[2]; } bm;
      bm.q[0] = *(const uint4*)(const void*)(brow + ks * 32);
      bm.q[1] = *(const uint4*)(const void*)(brow + ks * 32 + 16);
      acc = __builtin_amdgcn_wmma_f32_16x16x32_bf16(false, aT[ks], false, bm.v,
                                                    (short)0, acc, false, false);
    }
    int s = s0 + ml;
    float xxc = xx[b * T_N + s];
    bool valid = s < seg;
#pragma unroll
    for (int v = 0; v < 8; ++v) {
      float nd = valid ? (2.f * acc[v] - xxr[v] - xxc) : -3.4e38f;
      if (nd > bv0[v]) { bv2[v]=bv1[v]; bi2[v]=bi1[v]; bv1[v]=bv0[v]; bi1[v]=bi0[v]; bv0[v]=nd; bi0[v]=s; }
      else if (nd > bv1[v]) { bv2[v]=bv1[v]; bi2[v]=bi1[v]; bv1[v]=nd; bi1[v]=s; }
      else if (nd > bv2[v]) { bv2[v]=nd; bi2[v]=s; }
    }
  }
#pragma unroll
  for (int v = 0; v < 8; ++v) {
    int row = v + 8 * hi;
    lv[row][ml][0]=bv0[v]; lv[row][ml][1]=bv1[v]; lv[row][ml][2]=bv2[v];
    li[row][ml][0]=bi0[v]; li[row][ml][1]=bi1[v]; li[row][ml][2]=bi2[v];
  }
  __syncthreads();
  if (lane < 16) {
    float m0=-3.4e38f, m1=-3.4e38f, m2=-3.4e38f;
    int i0=0x7fffffff, i1=0x7fffffff, i2=0x7fffffff;
    for (int c = 0; c < 16; ++c)
      for (int j = 0; j < 3; ++j) {
        float vv = lv[lane][c][j]; int ii = li[lane][c][j];
        if (vv > m0 || (vv == m0 && ii < i0)) { m2=m1;i2=i1; m1=m0;i1=i0; m0=vv;i0=ii; }
        else if (vv > m1 || (vv == m1 && ii < i1)) { m2=m1;i2=i1; m1=vv;i1=ii; }
        else if (vv > m2 || (vv == m2 && ii < i2)) { m2=vv;i2=ii; }
      }
    int* o = idx_out + ((size_t)b * T_N + (t0 + lane)) * 3;
    o[0] = i0; o[1] = i1; o[2] = i2;
  }
}

// ---------- grouped conv1d k=3 groups=32 (t-major in/out), relu -> bf16 ----------
__global__ void gconv3_kernel(const float* __restrict__ in, const float* __restrict__ w,
                              const float* __restrict__ bias, unsigned short* __restrict__ outb) {
  int idx = blockIdx.x * blockDim.x + threadIdx.x;      // B*T*128, j fastest
  int j = idx & 127;
  int t = (idx >> 7) & (T_N - 1);
  int b = idx >> 18;
  const float* rowc = in + ((size_t)b * T_N + t) * 128 + (j >> 2) * 4;
  const float* wr = w + (size_t)j * 12;
  float s = bias[j];
#pragma unroll
  for (int ic = 0; ic < 4; ++ic) {
    if (t > 0)       s += wr[ic * 3 + 0] * rowc[ic - 128];   // t-1 row
    s += wr[ic * 3 + 1] * rowc[ic];
    if (t < T_N - 1) s += wr[ic * 3 + 2] * rowc[ic + 128];   // t+1 row
  }
  outb[idx] = f2bf(fmaxf(s, 0.f));
}

// ---------- grouped 1x1 conv groups=32 (t-major), relu -> bf16 ----------
__global__ void gconv1_kernel(const float* __restrict__ in, const float* __restrict__ w,
                              const float* __restrict__ bias, unsigned short* __restrict__ outb) {
  int idx = blockIdx.x * blockDim.x + threadIdx.x;      // B*T*128, j fastest
  int j = idx & 127;
  int t = (idx >> 7) & (T_N - 1);
  int b = idx >> 18;
  const float* rowc = in + ((size_t)b * T_N + t) * 128 + (j >> 2) * 4;
  const float* wr = w + (size_t)j * 4;
  float s = bias[j];
#pragma unroll
  for (int ic = 0; ic < 4; ++ic) s += wr[ic] * rowc[ic];
  outb[idx] = f2bf(fmaxf(s, 0.f));
}

// ---------- edge feature (graph branch, neighbor k): [neigh-center ; center] ----------
__global__ void ef_s_kernel(const float* __restrict__ x, const int* __restrict__ idx,
                            unsigned short* __restrict__ ef, int k) {
  int id = blockIdx.x * blockDim.x + threadIdx.x;       // B*T*512, c2 fastest
  int c2 = id & 511;
  int t  = (id >> 9) & (T_N - 1);
  int b  = id >> 20;
  float val;
  if (c2 < 256) {
    int nb = idx[((size_t)b * T_N + t) * 3 + k];
    val = x[((size_t)b * T_N + nb) * 256 + c2] - x[((size_t)b * T_N + t) * 256 + c2];
  } else {
    val = x[((size_t)b * T_N + t) * 256 + (c2 - 256)];
  }
  ef[id] = f2bf(val);
}

// ---------- fusion feature: [topic - x ; x] ----------
__global__ void ef_f_kernel(const float* __restrict__ x, const float* __restrict__ topicv,
                            unsigned short* __restrict__ ef) {
  int id = blockIdx.x * blockDim.x + threadIdx.x;       // B*T*512, c2 fastest
  int c2 = id & 511;
  int t  = (id >> 9) & (T_N - 1);
  int b  = id >> 20;
  float val;
  if (c2 < 256) val = topicv[b * 256 + c2] - x[((size_t)b * T_N + t) * 256 + c2];
  else          val = x[((size_t)b * T_N + t) * 256 + (c2 - 256)];
  ef[id] = f2bf(val);
}

// ---------- x = relu(t + s + f + x)  (all t-major) ----------
__global__ void combine_kernel(const float* __restrict__ yt, const float* __restrict__ sm,
                               const float* __restrict__ yf, float* __restrict__ x,
                               unsigned short* __restrict__ xb) {
  int id = blockIdx.x * blockDim.x + threadIdx.x;       // B*T*256
  float v = fmaxf(yt[id] + sm[id] + yf[id] + x[id], 0.f);
  x[id] = v;
  xb[id] = f2bf(v);
}

// ---------- final transpose: [B][T][256] -> [B][256][T] ----------
__global__ void transpose_kernel(const float* __restrict__ xt, float* __restrict__ out) {
  __shared__ float tile[32][33];
  int b  = blockIdx.z;
  int t0 = blockIdx.x * 32;
  int c0 = blockIdx.y * 32;
  int tx = threadIdx.x, ty = threadIdx.y;               // 32 x 8
#pragma unroll
  for (int i = 0; i < 32; i += 8)
    tile[ty + i][tx] = xt[((size_t)b * T_N + t0 + ty + i) * 256 + c0 + tx];
  __syncthreads();
#pragma unroll
  for (int i = 0; i < 32; i += 8)
    out[((size_t)b * 256 + c0 + ty + i) * T_N + t0 + tx] = tile[tx][ty + i];
}

// ============================== launch ==============================
extern "C" void kernel_launch(void* const* d_in, const int* in_sizes, int n_in,
                              void* d_out, int out_size, void* d_ws, size_t ws_size,
                              hipStream_t stream) {
  (void)in_sizes; (void)n_in; (void)out_size; (void)ws_size;
  const float* snip = (const float*)d_in[0];
  const float* te   = (const float*)d_in[1];
  const int*   seg  = (const int*)  d_in[2];
  const float* bb1w = (const float*)d_in[3];
  const float* bb1b = (const float*)d_in[4];
  const float* bbtw = (const float*)d_in[5];
  const float* bbtb = (const float*)d_in[6];
  const float* t1w = (const float*)d_in[7];  const float* t1b = (const float*)d_in[8];
  const float* t2w = (const float*)d_in[9];  const float* t2b = (const float*)d_in[10];
  const float* t3w = (const float*)d_in[11]; const float* t3b = (const float*)d_in[12];
  const float* s1w = (const float*)d_in[13]; const float* s1b = (const float*)d_in[14];
  const float* s2w = (const float*)d_in[15]; const float* s2b = (const float*)d_in[16];
  const float* s3w = (const float*)d_in[17]; const float* s3b = (const float*)d_in[18];
  const float* f1w = (const float*)d_in[19]; const float* f1b = (const float*)d_in[20];
  const float* f2w = (const float*)d_in[21]; const float* f2b = (const float*)d_in[22];
  const float* f3w = (const float*)d_in[23]; const float* f3b = (const float*)d_in[24];

  char* wsp = (char*)d_ws;
  size_t off = 0;
  auto alloc = [&](size_t bytes) -> void* {
    void* p = wsp + off;
    off += (bytes + 255) & ~(size_t)255;
    return p;
  };
  const size_t NBT = (size_t)B_N * 256 * T_N;           // 4M elements
  float*          x      = (float*)          alloc(NBT * 4);                   // [B][T][256]
  unsigned short* xb     = (unsigned short*) alloc(NBT * 2);                   // [B][T][256]
  float*          xxb    = (float*)          alloc((size_t)B_N * T_N * 4);
  int*            idxb   = (int*)            alloc((size_t)B_N * T_N * 3 * 4);
  float*          topicv = (float*)          alloc((size_t)B_N * 256 * 4);
  float*          bufW   = (float*)          alloc((size_t)B_N * T_N * 128 * 4);  // [B][T][128]
  unsigned short* bufWb  = (unsigned short*) alloc((size_t)B_N * T_N * 128 * 2);
  unsigned short* efb    = (unsigned short*) alloc((size_t)B_N * T_N * 512 * 2);  // [B][T][512]
  float*          bufT   = (float*)          alloc(NBT * 4);
  float*          smax   = (float*)          alloc(NBT * 4);
  float*          bufF   = (float*)          alloc(NBT * 4);
  unsigned short* wt1    = (unsigned short*) alloc(32768 * 2);
  unsigned short* wt3    = (unsigned short*) alloc(32768 * 2);
  unsigned short* ws1    = (unsigned short*) alloc(65536 * 2);
  unsigned short* ws3    = (unsigned short*) alloc(32768 * 2);
  unsigned short* wf1    = (unsigned short*) alloc(65536 * 2);
  unsigned short* wf3    = (unsigned short*) alloc(32768 * 2);

  dim3 blk(256);
  bb1_kernel<<<(unsigned)(NBT / 256), blk, 0, stream>>>(snip, bb1w, bb1b, x, xb);
  topic_kernel<<<(B_N * 256 + 255) / 256, blk, 0, stream>>>(te, bbtw, bbtb, topicv);

  for (int i = 0; i < 2; ++i) {
    cast_kernel<<<32768 / 256, blk, 0, stream>>>(t1w + i * 32768, wt1, 32768);
    cast_kernel<<<32768 / 256, blk, 0, stream>>>(t3w + i * 32768, wt3, 32768);
    cast_kernel<<<65536 / 256, blk, 0, stream>>>(s1w + i * 65536, ws1, 65536);
    cast_kernel<<<32768 / 256, blk, 0, stream>>>(s3w + i * 32768, ws3, 32768);
    cast_kernel<<<65536 / 256, blk, 0, stream>>>(f1w + i * 65536, wf1, 65536);
    cast_kernel<<<32768 / 256, blk, 0, stream>>>(f3w + i * 32768, wf3, 32768);

    // --- temporal branch: t1 (256->128, relu) -> t2 grouped k3 (relu->bf16) -> t3 (128->256)
    gemm_bf16_kernel<<<dim3(T_N / 16, 2, B_N), dim3(32), 0, stream>>>(wt1, xb, t1b + i * 128, bufW, 128, 256, 1);
    gconv3_kernel<<<(unsigned)((size_t)B_N * T_N * 128 / 256), blk, 0, stream>>>(bufW, t2w + i * 1536, t2b + i * 128, bufWb);
    gemm_bf16_kernel<<<dim3(T_N / 16, 4, B_N), dim3(32), 0, stream>>>(wt3, bufWb, t3b + i * 256, bufT, 256, 128, 0);

    // --- knn
    xx_kernel<<<(B_N * T_N) / 256, blk, 0, stream>>>(x, xxb);
    knn_kernel<<<dim3(T_N / 16, B_N), dim3(32), 0, stream>>>(xb, xxb, seg, idxb);

    // --- graph branch, per neighbor, max-reduced into smax (k=0 stores, k>0 max-acc)
    for (int k = 0; k < 3; ++k) {
      ef_s_kernel<<<(unsigned)((size_t)B_N * T_N * 512 / 256), blk, 0, stream>>>(x, idxb, efb, k);
      gemm_bf16_kernel<<<dim3(T_N / 16, 2, B_N), dim3(32), 0, stream>>>(ws1, efb, s1b + i * 128, bufW, 128, 512, 1);
      gconv1_kernel<<<(unsigned)((size_t)B_N * T_N * 128 / 256), blk, 0, stream>>>(bufW, s2w + i * 512, s2b + i * 128, bufWb);
      gemm_bf16_kernel<<<dim3(T_N / 16, 4, B_N), dim3(32), 0, stream>>>(ws3, bufWb, s3b + i * 256, smax, 256, 128, (k == 0) ? 0 : 2);
    }

    // --- fusion branch
    ef_f_kernel<<<(unsigned)((size_t)B_N * T_N * 512 / 256), blk, 0, stream>>>(x, topicv, efb);
    gemm_bf16_kernel<<<dim3(T_N / 16, 2, B_N), dim3(32), 0, stream>>>(wf1, efb, f1b + i * 128, bufW, 128, 512, 1);
    gconv1_kernel<<<(unsigned)((size_t)B_N * T_N * 128 / 256), blk, 0, stream>>>(bufW, f2w + i * 512, f2b + i * 128, bufWb);
    gemm_bf16_kernel<<<dim3(T_N / 16, 4, B_N), dim3(32), 0, stream>>>(wf3, bufWb, f3b + i * 256, bufF, 256, 128, 0);

    combine_kernel<<<(unsigned)(NBT / 256), blk, 0, stream>>>(bufT, smax, bufF, x, xb);
  }

  // final layout change [B][T][256] -> [B][256][T]
  transpose_kernel<<<dim3(T_N / 32, 256 / 32, B_N), dim3(32, 8), 0, stream>>>(x, (float*)d_out);
}
